// winconv2d_30889404793385
// MI455X (gfx1250) — matrix operand order repaired and checked by reading, not compiled, not run on hardware
//
#include <hip/hip_runtime.h>

typedef __attribute__((ext_vector_type(16))) _Float16 v16h;
typedef __attribute__((ext_vector_type(8)))  float    v8f;

#define CIN   128
#define COUT  256
#define HW    64
#define NTILE 1024   // 32x32 Winograd tiles per image

// ---------------------------------------------------------------------------
// K1: abs-max reduction of weight -> ws[0] (float bits, atomicMax on uint
// works because |w| >= 0 so the uint ordering matches the float ordering)
// ---------------------------------------------------------------------------
__global__ void absmax_kernel(const float* __restrict__ w, unsigned* smax, int n) {
  unsigned m = 0u;
  for (int i = blockIdx.x * blockDim.x + threadIdx.x; i < n;
       i += gridDim.x * blockDim.x) {
    unsigned u = __float_as_uint(fabsf(w[i]));
    m = (u > m) ? u : m;
  }
  atomicMax(smax, m);
}

// ---------------------------------------------------------------------------
// K2: quant-dequant weight and Winograd weight transform U = G wq G^T.
// Store f16 as U[p][o][c] so B-fragments are contiguous per lane.
// ---------------------------------------------------------------------------
__global__ void wtrans_kernel(const float* __restrict__ w,
                              const unsigned* __restrict__ smax,
                              _Float16* __restrict__ U) {
  int id = blockIdx.x * blockDim.x + threadIdx.x;
  if (id >= COUT * CIN) return;
  int o = id / CIN;
  int c = id % CIN;
  float scale = __uint_as_float(*smax) * (1.0f / 127.0f);
  float inv = (scale > 0.0f) ? (1.0f / scale) : 0.0f;
  const float* wp = w + (size_t)id * 9;
  float q[3][3];
#pragma unroll
  for (int r = 0; r < 3; ++r)
#pragma unroll
    for (int s = 0; s < 3; ++s)
      q[r][s] = rintf(wp[r * 3 + s] * inv) * scale;  // RNE, matches jnp.round
  // t = G * q  (4x3)
  float t[4][3];
#pragma unroll
  for (int j = 0; j < 3; ++j) {
    t[0][j] = q[0][j];
    t[1][j] = 0.5f * (q[0][j] + q[1][j] + q[2][j]);
    t[2][j] = 0.5f * (q[0][j] - q[1][j] + q[2][j]);
    t[3][j] = q[2][j];
  }
  // U = t * G^T  (4x4)
#pragma unroll
  for (int i = 0; i < 4; ++i) {
    float u0 = t[i][0];
    float u1 = 0.5f * (t[i][0] + t[i][1] + t[i][2]);
    float u2 = 0.5f * (t[i][0] - t[i][1] + t[i][2]);
    float u3 = t[i][2];
    U[((size_t)((i * 4 + 0) * COUT + o)) * CIN + c] = (_Float16)u0;
    U[((size_t)((i * 4 + 1) * COUT + o)) * CIN + c] = (_Float16)u1;
    U[((size_t)((i * 4 + 2) * COUT + o)) * CIN + c] = (_Float16)u2;
    U[((size_t)((i * 4 + 3) * COUT + o)) * CIN + c] = (_Float16)u3;
  }
}

// ---------------------------------------------------------------------------
// K3: fused input-transform + Winograd-domain GEMM (WMMA) + output-transform.
// Block: 256 threads = 8 waves. Block covers 16 tile-rows x 128 out channels;
// wave covers one 16-wide o chunk and all 16 Winograd positions.
// ---------------------------------------------------------------------------
__global__ __launch_bounds__(256, 1) void winconv_main(
    const float* __restrict__ x, const _Float16* __restrict__ U,
    float* __restrict__ out) {
  // LDS: X staging for one K-chunk: [p][row][cl] f16, 16*16*32*2 = 16 KB
  __shared__ __align__(16) _Float16 Xs[16 * 16 * 32];

  const int tid = threadIdx.x;
  const int lane = tid & 31;
  const int wave = tid >> 5;
  const int laneHi = lane >> 4;   // 0/1: selects K half per ISA layouts
  const int laneLo = lane & 15;   // A: row M;  B/D: column N

  const int n0 = (blockIdx.x >> 1) * 16;                 // tile-row block
  const int o0 = (blockIdx.x & 1) * 128 + wave * 16;     // out-channel chunk
  const int b = n0 / NTILE;  // all 16 rows share one batch (16 | 1024)

  v8f acc[16];
#pragma unroll
  for (int p = 0; p < 16; ++p) acc[p] = v8f{0.f, 0.f, 0.f, 0.f, 0.f, 0.f, 0.f, 0.f};

  for (int c0 = 0; c0 < CIN; c0 += 32) {
    __syncthreads();  // previous chunk fully consumed before overwrite
    // ---- Phase 1: X = BT d BT^T for 16 rows x 32 channels -> LDS (f16)
#pragma unroll
    for (int it = 0; it < 2; ++it) {
      int id = tid + it * 256;      // 0..511
      int cl = id & 31;
      int row = id >> 5;            // 0..15
      int n = n0 + row;
      int t = n & (NTILE - 1);
      int th = t >> 5, tw = t & 31;
      int ih0 = th * 2 - 1, iw0 = tw * 2 - 1;  // includes PAD=1
      const float* xp = x + ((size_t)(b * CIN + (c0 + cl)) * HW) * HW;
      float d[4][4];
#pragma unroll
      for (int r = 0; r < 4; ++r) {
        int ih = ih0 + r;
#pragma unroll
        for (int s = 0; s < 4; ++s) {
          int iw = iw0 + s;
          bool ok = ((unsigned)ih < HW) && ((unsigned)iw < HW);
          d[r][s] = ok ? xp[ih * HW + iw] : 0.0f;
        }
      }
      float tm[4][4];  // BT * d
#pragma unroll
      for (int j = 0; j < 4; ++j) {
        tm[0][j] = d[0][j] - d[2][j];
        tm[1][j] = d[1][j] + d[2][j];
        tm[2][j] = d[2][j] - d[1][j];
        tm[3][j] = d[1][j] - d[3][j];
      }
#pragma unroll
      for (int i = 0; i < 4; ++i) {  // (BT d) * BT^T
        float X0 = tm[i][0] - tm[i][2];
        float X1 = tm[i][1] + tm[i][2];
        float X2 = tm[i][2] - tm[i][1];
        float X3 = tm[i][1] - tm[i][3];
        Xs[((i * 4 + 0) * 16 + row) * 32 + cl] = (_Float16)X0;
        Xs[((i * 4 + 1) * 16 + row) * 32 + cl] = (_Float16)X1;
        Xs[((i * 4 + 2) * 16 + row) * 32 + cl] = (_Float16)X2;
        Xs[((i * 4 + 3) * 16 + row) * 32 + cl] = (_Float16)X3;
      }
    }
    __syncthreads();
    // ---- Phase 2: one 16x16x32 WMMA per Winograd position
#pragma unroll
    for (int p = 0; p < 16; ++p) {
      union { v16h v; uint4 q[2]; } af, bf;
      // A fragment (ISA 16-bit A 16x32): lane row = laneLo;
      // halves 0..7 -> K = h + 8*laneHi ; halves 8..15 -> K = h+8 + 8*laneHi + 8
      const _Float16* abase = &Xs[(p * 16 + laneLo) * 32];
      af.q[0] = *(const uint4*)(abase + 8 * laneHi);
      af.q[1] = *(const uint4*)(abase + 16 + 8 * laneHi);
      // B fragment (32x16): lane col = laneLo; same K order as A
      const _Float16* bbase =
          U + ((size_t)(p * COUT + (o0 + laneLo))) * CIN + c0 + 8 * laneHi;
      bf.q[0] = *(const uint4*)(bbase);
      bf.q[1] = *(const uint4*)(bbase + 16);
      acc[p] = __builtin_amdgcn_wmma_f32_16x16x32_f16(
          false, af.v, false, bf.v, (short)0, acc[p], false, false);
    }
  }

  // ---- Output transform Y = AT M AT^T, per lane: col o fixed, 8 rows
  const int o = o0 + laneLo;
#pragma unroll
  for (int e = 0; e < 8; ++e) {
    int row = e + 8 * laneHi;      // D layout: VGPR e holds M = e (+8 hi half)
    int n = n0 + row;
    int t = n & (NTILE - 1);
    int th = t >> 5, tw = t & 31;
    float M[4][4];
#pragma unroll
    for (int p = 0; p < 16; ++p) M[p >> 2][p & 3] = acc[p][e];
    float a0[4], a1[4];
#pragma unroll
    for (int k = 0; k < 4; ++k) {
      a0[k] = M[0][k] + M[1][k] + M[2][k];
      a1[k] = M[1][k] - M[2][k] - M[3][k];
    }
    float y00 = a0[0] + a0[1] + a0[2];
    float y01 = a0[1] - a0[2] - a0[3];
    float y10 = a1[0] + a1[1] + a1[2];
    float y11 = a1[1] - a1[2] - a1[3];
    float* op = out + (((size_t)(b * COUT + o) * HW) + th * 2) * HW + tw * 2;
    *(float2*)(op)      = make_float2(y00, y01);
    *(float2*)(op + HW) = make_float2(y10, y11);
  }
}

// ---------------------------------------------------------------------------
extern "C" void kernel_launch(void* const* d_in, const int* in_sizes, int n_in,
                              void* d_out, int out_size, void* d_ws, size_t ws_size,
                              hipStream_t stream) {
  const float* x = (const float*)d_in[0];       // [8,128,64,64]
  const float* w = (const float*)d_in[1];       // [256,128,3,3]
  float* out = (float*)d_out;                   // [8,256,64,64]

  unsigned* smax = (unsigned*)d_ws;                       // 4 bytes
  _Float16* U = (_Float16*)((char*)d_ws + 256);           // 16*256*128 f16 = 1 MB

  int nw = in_sizes[1];  // 294912
  hipMemsetAsync(smax, 0, sizeof(unsigned), stream);
  absmax_kernel<<<288, 256, 0, stream>>>(w, smax, nw);
  wtrans_kernel<<<(COUT * CIN + 255) / 256, 256, 0, stream>>>(w, smax, U);
  // 512 row-blocks (8192 tile-rows / 16) x 2 o-halves = 1024 workgroups
  winconv_main<<<1024, 256, 0, stream>>>(x, U, out);
  (void)n_in; (void)out_size; (void)ws_size;
}
